// GRIL_652835029743
// MI455X (gfx1250) — matrix operand order repaired and checked by reading, not compiled, not run on hardware
//
#include <hip/hip_runtime.h>
#include <math.h>

// ---------------------------------------------------------------------------
// GRIL on MI455X (gfx1250): all contractions via v_wmma_f32_16x16x32_bf16.
// B=16, N=500 (padded 512), T=128, H=64, F_IN=1.
// ---------------------------------------------------------------------------

#define NP 512                      // padded node dim
typedef __attribute__((ext_vector_type(16))) __bf16 v16bf;
typedef __attribute__((ext_vector_type(8)))  __bf16 v8bf;
typedef __attribute__((ext_vector_type(8)))  float  v8f;

// ---------------------------------------------------------------------------
// Generic batched WMMA GEMM:  C[b] (Mv x 512) = A[b](M x K) * B[b](K x 512)
// A,B bf16 row-major; C written f32 and/or bf16 with fused bias+activation.
// Workgroup: 256 thr = 8 waves, tile 32(M) x 512(N); each wave 32x64 (8 WMMA
// per 32-K chunk). LDS double-buffered: stage chunk k+1 while computing k.
// ---------------------------------------------------------------------------
__global__ __launch_bounds__(256)
void wmma_gemm(const __bf16* __restrict__ A, long asb, int lda,
               const __bf16* __restrict__ B, long bsb, int ldb,
               int K, int Mv,
               const float* __restrict__ bias,
               int act, const float* __restrict__ act_p,
               float* __restrict__ Cf, long cfs,
               __bf16* __restrict__ Cb, long cbs)
{
    __shared__ __align__(16) __bf16 sA[2][32][32];    // [buf][m][k]
    __shared__ __align__(16) __bf16 sB[2][NP][32];    // [buf][n][k] (transposed)

    const int tid  = threadIdx.x;
    const int lane = tid & 31;
    const int wn   = tid >> 5;                 // wave -> N sub-tile (0..7)
    const int lh   = lane >> 4;                // lane half (0/1)
    const int ll   = lane & 15;
    const int m0   = blockIdx.y * 32;
    const int b    = blockIdx.z;

    const __bf16* Ag = A + (long)b * asb;
    const __bf16* Bg = B + (long)b * bsb;

    v8f acc[2][4];
#pragma unroll
    for (int ma = 0; ma < 2; ++ma)
#pragma unroll
        for (int s = 0; s < 4; ++s)
            acc[ma][s] = (v8f){0.f,0.f,0.f,0.f,0.f,0.f,0.f,0.f};

    // ---- cooperative staging of one 32-K chunk into LDS buffer `buf` ----
    auto stage = [&](int k0, int buf) {
        // A tile 32x32: 256 thr x 4 bf16 (8B) each, coalesced
        {
            const int m  = tid >> 3;
            const int kg = (tid & 7) << 2;
            *(unsigned long long*)&sA[buf][m][kg] =
                *(const unsigned long long*)(Ag + (long)(m0 + m) * lda + k0 + kg);
        }
        // B tile 32k x 512n, transposed in registers (8k x 2n blocks),
        // written with two ds_store_b128 per block.
#pragma unroll
        for (int j = 0; j < 4; ++j) {
            const int idx = tid + 256 * j;
            const int k8  = (idx >> 8) << 3;     // 0,8,16,24
            const int n2  = (idx & 255) << 1;    // 0..510 step 2
            v8bf lo, hi;
#pragma unroll
            for (int kk = 0; kk < 8; ++kk) {
                union { unsigned int u; __bf16 h[2]; } cv;
                cv.u = *(const unsigned int*)(Bg + (long)(k0 + k8 + kk) * ldb + n2);
                lo[kk] = cv.h[0];
                hi[kk] = cv.h[1];
            }
            *(v8bf*)&sB[buf][n2][k8]     = lo;
            *(v8bf*)&sB[buf][n2 + 1][k8] = hi;
        }
    };

    stage(0, 0);
    __syncthreads();

    int cur = 0;
    for (int k0 = 0; k0 < K; k0 += 32) {
        const int nx = k0 + 32;
        if (nx < K) stage(nx, cur ^ 1);          // overlap staging with WMMA
        if (nx + 32 < K)                          // prefetch 2 chunks ahead
            __builtin_prefetch(Bg + (long)(nx + 32 + (tid >> 3)) * ldb +
                               ((tid & 7) << 6), 0, 1);

        // A fragments (16x32 each): lane ll = row; K halves 0-15 vs 16-31 (contiguous)
        v16bf a0, a1;
        ((v8bf*)&a0)[0] = *(const v8bf*)&sA[cur][ll][lh * 16];
        ((v8bf*)&a0)[1] = *(const v8bf*)&sA[cur][ll][lh * 16 + 8];
        ((v8bf*)&a1)[0] = *(const v8bf*)&sA[cur][16 + ll][lh * 16];
        ((v8bf*)&a1)[1] = *(const v8bf*)&sA[cur][16 + ll][lh * 16 + 8];

#pragma unroll
        for (int s = 0; s < 4; ++s) {
            // B fragment (32x16): lane ll = col, 16 contiguous K per lane half
            v16bf bfrag = *(const v16bf*)&sB[cur][wn * 64 + s * 16 + ll][lh * 16];
            acc[0][s] = __builtin_amdgcn_wmma_f32_16x16x32_bf16(
                false, a0, false, bfrag, (short)0, acc[0][s], false, false);
            acc[1][s] = __builtin_amdgcn_wmma_f32_16x16x32_bf16(
                false, a1, false, bfrag, (short)0, acc[1][s], false, false);
        }
        __syncthreads();                          // one barrier per chunk
        cur ^= 1;
    }

    const float alpha = (act == 3 && act_p) ? *act_p : 0.f;
#pragma unroll
    for (int ma = 0; ma < 2; ++ma) {
#pragma unroll
        for (int s = 0; s < 4; ++s) {
            const int n = wn * 64 + s * 16 + ll;
#pragma unroll
            for (int r = 0; r < 8; ++r) {
                const int m = m0 + ma * 16 + lh * 8 + r;
                if (m < Mv) {
                    float v = acc[ma][s][r];
                    if (bias) v += bias[m];
                    if      (act == 1) v = 1.f / (1.f + __expf(-v));
                    else if (act == 2) v = tanhf(v);
                    else if (act == 3) v = (v >= 0.f) ? v : alpha * v;
                    if (Cf) Cf[(long)b * cfs + (long)m * NP + n] = v;
                    if (Cb) Cb[(long)b * cbs + (long)m * NP + n] = (__bf16)v;
                }
            }
        }
    }
}

// ---------------------------------------------------------------------------
// Setup / pointwise kernels
// ---------------------------------------------------------------------------
__global__ void k_zero(unsigned int* __restrict__ p, long nwords)
{
    long i = (long)blockIdx.x * blockDim.x + threadIdx.x;
    long stride = (long)gridDim.x * blockDim.x;
    for (; i < nwords; i += stride) p[i] = 0u;
}

// AfT[k][n] = af[n][k], bf16, zero-padded to 512x512
__global__ void k_adjT(__bf16* __restrict__ dst, const float* __restrict__ src)
{
    long i = (long)blockIdx.x * blockDim.x + threadIdx.x;
    if (i >= (long)NP * NP) return;
    int k = (int)(i >> 9), n = (int)(i & 511);
    float v = (k < 500 && n < 500) ? src[(long)n * 500 + k] : 0.f;
    dst[i] = (__bf16)v;
}

// dst[Mp x Kp] bf16 <- src[M x K] f32, zero pad
__global__ void k_packw(__bf16* __restrict__ dst, const float* __restrict__ src,
                        int M, int K, int Mp, int Kp)
{
    int i = blockIdx.x * blockDim.x + threadIdx.x;
    if (i >= Mp * Kp) return;
    int m = i / Kp, k = i % Kp;
    dst[i] = (m < M && k < K) ? (__bf16)src[(long)m * K + k] : (__bf16)0.f;
}

// stage-1 select: x1 = mask ? x : xhat1 ; emit pred, cat1 row0/1, mf rows
__global__ void k_p1(const float* __restrict__ x, const int* __restrict__ mask,
                     const float* __restrict__ xhat1, float* __restrict__ x1f,
                     __bf16* __restrict__ cat1, __bf16* __restrict__ catg,
                     __bf16* __restrict__ catc, float* __restrict__ pred_out, int t)
{
    int i = blockIdx.x * blockDim.x + threadIdx.x;
    if (i >= 16 * 500) return;
    int b = i / 500, n = i % 500;
    long xo = (long)b * 500 * 128 + (long)n * 128 + t;
    float xh = xhat1[(long)b * NP + n];
    int   m  = mask[xo];
    float x1 = m ? x[xo] : xh;
    float mf = (float)m;
    x1f[(long)b * NP + n] = x1;
    cat1[(long)b * 96 * NP + n]       = (__bf16)x1;  // row 0
    cat1[(long)b * 96 * NP + NP + n]  = (__bf16)mf;  // row 1
    catg[(long)b * 384 * NP + NP + n] = (__bf16)mf;  // row 1
    catc[(long)b * 384 * NP + NP + n] = (__bf16)mf;  // row 1
    pred_out[xo] = xh;
}

// stage-2 select: x2 = mask ? x1 : xhat2 ; emit imp, xg/xgc row0
__global__ void k_p3(const int* __restrict__ mask, const float* __restrict__ x1f,
                     const float* __restrict__ xhat2, __bf16* __restrict__ catg,
                     __bf16* __restrict__ catc, float* __restrict__ imp_out, int t)
{
    int i = blockIdx.x * blockDim.x + threadIdx.x;
    if (i >= 16 * 500) return;
    int b = i / 500, n = i % 500;
    long xo = (long)b * 500 * 128 + (long)n * 128 + t;
    float xh2 = xhat2[(long)b * NP + n];
    float x2  = mask[xo] ? x1f[(long)b * NP + n] : xh2;
    catg[(long)b * 384 * NP + n] = (__bf16)x2;       // row 0
    catc[(long)b * 384 * NP + n] = (__bf16)x2;       // row 0
    imp_out[xo] = xh2;
}

// r*h into catc rows 2..65 ; write reps output ([lo ; h]) for this t
__global__ void k_p4(const float* __restrict__ hf, const float* __restrict__ rf,
                     const float* __restrict__ lof, __bf16* __restrict__ catc,
                     float* __restrict__ reps_out, int t)
{
    int i = blockIdx.x * blockDim.x + threadIdx.x;
    if (i >= 16 * 64 * 500) return;
    int b = i / (64 * 500);
    int r = i % (64 * 500);
    int c = r / 500, n = r % 500;
    long ho = (long)b * 64 * NP + (long)c * NP + n;
    float h = hf[ho];
    catc[(long)b * 384 * NP + (long)(2 + c) * NP + n] = (__bf16)(rf[ho] * h);
    long ro = (long)b * 128 * 500 * 128 + (long)c * 500 * 128 + (long)n * 128 + t;
    reps_out[ro] = lof[ho];
    reps_out[ro + (long)64 * 500 * 128] = h;
}

// GRU update h = u*h + (1-u)*c ; broadcast bf16 h into all concat buffers
__global__ void k_p5(float* __restrict__ hf, __bf16* __restrict__ hbf,
                     const float* __restrict__ uf, const float* __restrict__ cf,
                     __bf16* __restrict__ cat1, __bf16* __restrict__ cat_lo,
                     __bf16* __restrict__ cat_ro, __bf16* __restrict__ catg)
{
    int i = blockIdx.x * blockDim.x + threadIdx.x;
    if (i >= 16 * 64 * NP) return;
    int b = i >> 15;           // /(64*512)
    int c = (i >> 9) & 63;
    int n = i & 511;
    long ho = (long)b * 64 * NP + (long)c * NP + n;
    float u = uf[ho];
    float h = u * hf[ho] + (1.f - u) * cf[ho];
    hf[ho] = h;
    __bf16 hb = (__bf16)h;
    hbf[ho] = hb;
    cat1  [(long)b * 96  * NP + (long)(2 + c)  * NP + n] = hb;
    cat_lo[(long)b * 128 * NP + (long)(64 + c) * NP + n] = hb;
    cat_ro[(long)b * 128 * NP + (long)(64 + c) * NP + n] = hb;
    catg  [(long)b * 384 * NP + (long)(2 + c)  * NP + n] = hb;
}

// ---------------------------------------------------------------------------
// Host launcher
// ---------------------------------------------------------------------------
extern "C" void kernel_launch(void* const* d_in, const int* in_sizes, int n_in,
                              void* d_out, int out_size, void* d_ws, size_t ws_size,
                              hipStream_t stream)
{
    const float* x    = (const float*)d_in[0];
    const float* af   = (const float*)d_in[1];
    const float* ab   = (const float*)d_in[2];
    const int*   mask = (const int*)  d_in[3];
    const float* Wr   = (const float*)d_in[4];   const float* br   = (const float*)d_in[5];
    const float* Wu   = (const float*)d_in[6];   const float* bu   = (const float*)d_in[7];
    const float* Wc   = (const float*)d_in[8];   const float* bc   = (const float*)d_in[9];
    const float* W_fs = (const float*)d_in[10];  const float* b_fs = (const float*)d_in[11];
    const float* W_li = (const float*)d_in[12];  const float* b_li = (const float*)d_in[13];
    const float* W_gc = (const float*)d_in[14];  const float* b_gc = (const float*)d_in[15];
    const float* W_lo = (const float*)d_in[16];  const float* b_lo = (const float*)d_in[17];
    const float* W_ro = (const float*)d_in[18];  const float* b_ro = (const float*)d_in[19];
    const float* prelu_a = (const float*)d_in[20];

    float* imp_out  = (float*)d_out;
    float* pred_out = imp_out + (long)16 * 500 * 128;
    float* reps_out = imp_out + (long)2 * 16 * 500 * 128;

    // ---- workspace layout (256B aligned) ----
    size_t off = 0;
    auto alloc = [&](size_t bytes) -> char* {
        char* p = (char*)d_ws + off;
        off += (bytes + 255) & ~(size_t)255;
        return p;
    };
    __bf16* AfT   = (__bf16*)alloc((size_t)NP * NP * 2);
    __bf16* AbT   = (__bf16*)alloc((size_t)NP * NP * 2);
    __bf16* Wfs_p = (__bf16*)alloc(32  * 64  * 2);
    __bf16* Wli_p = (__bf16*)alloc(64  * 96  * 2);
    __bf16* Wgc_p = (__bf16*)alloc(64  * 128 * 2);
    __bf16* Wlo_p = (__bf16*)alloc(64  * 128 * 2);
    __bf16* Wro_p = (__bf16*)alloc(32  * 128 * 2);
    __bf16* Wr_p  = (__bf16*)alloc(64  * 352 * 2);
    __bf16* Wu_p  = (__bf16*)alloc(64  * 352 * 2);
    __bf16* Wc_p  = (__bf16*)alloc(64  * 352 * 2);
    float * h_f   = (float *)alloc((size_t)16 * 64 * NP * 4);
    __bf16* h_bf  = (__bf16*)alloc((size_t)16 * 64 * NP * 2);
    float * xhat1 = (float *)alloc((size_t)16 * NP * 4);
    float * x1f   = (float *)alloc((size_t)16 * NP * 4);
    float * xhat2 = (float *)alloc((size_t)16 * NP * 4);
    __bf16* cat1  = (__bf16*)alloc((size_t)16 * 96  * NP * 2);
    __bf16* xin_b = (__bf16*)alloc((size_t)16 * 64  * NP * 2);
    __bf16* cat_gc= (__bf16*)alloc((size_t)16 * 128 * NP * 2);
    __bf16* cat_lo= (__bf16*)alloc((size_t)16 * 128 * NP * 2);
    __bf16* cat_ro= (__bf16*)alloc((size_t)16 * 128 * NP * 2);
    float * lo_f  = (float *)alloc((size_t)16 * 64 * NP * 4);
    float * r_f   = (float *)alloc((size_t)16 * 64 * NP * 4);
    float * u_f   = (float *)alloc((size_t)16 * 64 * NP * 4);
    float * c_f   = (float *)alloc((size_t)16 * 64 * NP * 4);
    __bf16* catg  = (__bf16*)alloc((size_t)16 * 384 * NP * 2);
    __bf16* catc  = (__bf16*)alloc((size_t)16 * 384 * NP * 2);
    const size_t total = off;

    // ---- one-time setup (zero pads / h, pack operands to bf16) ----
    k_zero<<<4096, 256, 0, stream>>>((unsigned int*)d_ws, (long)(total / 4));
    k_adjT<<<(NP * NP + 255) / 256, 256, 0, stream>>>(AfT, af);
    k_adjT<<<(NP * NP + 255) / 256, 256, 0, stream>>>(AbT, ab);
    auto pack = [&](__bf16* dst, const float* src, int M, int K, int Mp, int Kp) {
        k_packw<<<(Mp * Kp + 255) / 256, 256, 0, stream>>>(dst, src, M, K, Mp, Kp);
    };
    pack(Wfs_p, W_fs, 1, 64, 32, 64);
    pack(Wli_p, W_li, 64, 66, 64, 96);
    pack(Wgc_p, W_gc, 64, 128, 64, 128);
    pack(Wlo_p, W_lo, 64, 128, 64, 128);
    pack(Wro_p, W_ro, 1, 128, 32, 128);
    pack(Wr_p,  Wr,   64, 330, 64, 352);
    pack(Wu_p,  Wu,   64, 330, 64, 352);
    pack(Wc_p,  Wc,   64, 330, 64, 352);

    auto gemm = [&](const __bf16* A, long asb, int lda,
                    const __bf16* B, long bsb, int ldb, int K, int Mv,
                    const float* bias, int act, const float* actp,
                    float* Cf, long cfs, __bf16* Cb, long cbs) {
        dim3 g(1, (Mv + 31) / 32, 16);
        wmma_gemm<<<g, 256, 0, stream>>>(A, asb, lda, B, bsb, ldb, K, Mv,
                                         bias, act, actp, Cf, cfs, Cb, cbs);
    };

    const long S64 = 64L * NP, S96 = 96L * NP, S128 = 128L * NP, S384 = 384L * NP;

    for (int t = 0; t < 128; ++t) {
        // xs_hat_1 = W_fs . h
        gemm(Wfs_p, 0, 64, h_bf, S64, NP, 64, 1, b_fs, 0, nullptr,
             xhat1, NP, nullptr, 0);
        k_p1<<<32, 256, 0, stream>>>(x, mask, xhat1, x1f, cat1, catg, catc, pred_out, t);
        // x_in = W_li . [x1; mf; h]
        gemm(Wli_p, 0, 96, cat1, S96, NP, 96, 64, b_li, 0, nullptr,
             nullptr, 0, xin_b, S64);
        // decoder diffusion: x_in @ Af^T , x_in @ Ab^T  -> cat_gc[0:64],[64:128]
        gemm(xin_b, S64, NP, AfT, 0, NP, NP, 64, nullptr, 0, nullptr,
             nullptr, 0, cat_gc, S128);
        gemm(xin_b, S64, NP, AbT, 0, NP, NP, 64, nullptr, 0, nullptr,
             nullptr, 0, cat_gc + 64 * NP, S128);
        // gc_out = W_gc . cat_gc  -> cat_lo[0:64]
        gemm(Wgc_p, 0, 128, cat_gc, S128, NP, 128, 64, b_gc, 0, nullptr,
             nullptr, 0, cat_lo, S128);
        // lo = prelu(W_lo . [gc_out; h])  -> lo_f + cat_ro[0:64]
        gemm(Wlo_p, 0, 128, cat_lo, S128, NP, 128, 64, b_lo, 3, prelu_a,
             lo_f, S64, cat_ro, S128);
        // xs_hat_2 = W_ro . [lo; h]
        gemm(Wro_p, 0, 128, cat_ro, S128, NP, 128, 1, b_ro, 0, nullptr,
             xhat2, NP, nullptr, 0);
        k_p3<<<32, 256, 0, stream>>>(mask, x1f, xhat2, catg, catc, imp_out, t);
        // order-2 diffusion on xg = catg[0:66] -> blocks at rows 66/132/198/264
        gemm(catg,            S384, NP, AfT, 0, NP, NP, 66, nullptr, 0, nullptr,
             nullptr, 0, catg + 66 * NP,  S384);
        gemm(catg + 66 * NP,  S384, NP, AfT, 0, NP, NP, 66, nullptr, 0, nullptr,
             nullptr, 0, catg + 132 * NP, S384);
        gemm(catg,            S384, NP, AbT, 0, NP, NP, 66, nullptr, 0, nullptr,
             nullptr, 0, catg + 198 * NP, S384);
        gemm(catg + 198 * NP, S384, NP, AbT, 0, NP, NP, 66, nullptr, 0, nullptr,
             nullptr, 0, catg + 264 * NP, S384);
        // r, u gates (K=352 incl. zero pad)
        gemm(Wr_p, 0, 352, catg, S384, NP, 352, 64, br, 1, nullptr,
             r_f, S64, nullptr, 0);
        gemm(Wu_p, 0, 352, catg, S384, NP, 352, 64, bu, 1, nullptr,
             u_f, S64, nullptr, 0);
        k_p4<<<(16 * 64 * 500 + 255) / 256, 256, 0, stream>>>(h_f, r_f, lo_f, catc, reps_out, t);
        // order-2 diffusion on xgc = [x2; mf; r*h]
        gemm(catc,            S384, NP, AfT, 0, NP, NP, 66, nullptr, 0, nullptr,
             nullptr, 0, catc + 66 * NP,  S384);
        gemm(catc + 66 * NP,  S384, NP, AfT, 0, NP, NP, 66, nullptr, 0, nullptr,
             nullptr, 0, catc + 132 * NP, S384);
        gemm(catc,            S384, NP, AbT, 0, NP, NP, 66, nullptr, 0, nullptr,
             nullptr, 0, catc + 198 * NP, S384);
        gemm(catc + 198 * NP, S384, NP, AbT, 0, NP, NP, 66, nullptr, 0, nullptr,
             nullptr, 0, catc + 264 * NP, S384);
        // candidate c = tanh(Wc . catc)
        gemm(Wc_p, 0, 352, catc, S384, NP, 352, 64, bc, 2, nullptr,
             c_f, S64, nullptr, 0);
        // h update + broadcast bf16 h into concat buffers for next step
        k_p5<<<(16 * 64 * NP) / 256, 256, 0, stream>>>(h_f, h_bf, u_f, c_f,
                                                       cat1, cat_lo, cat_ro, catg);
    }
    (void)in_sizes; (void)n_in; (void)out_size; (void)ws_size;
}
